// HybridGraphConditionedTwoWayBlock_15161234555373
// MI455X (gfx1250) — matrix operand order repaired and checked by reading, not compiled, not run on hardware
//
#include <hip/hip_runtime.h>
#include <hip/hip_bf16.h>

// ---------------------------------------------------------------------------
// CDNA5 (gfx1250) implementation: all GEMMs via v_wmma_f32_16x16x32_bf16.
// Every GEMM is expressed as  C(MxN) = A(MxK) @ Bt(NxK)^T  so that one
// fragment-load pattern (two 16B chunks per lane) serves A and B everywhere.
// ---------------------------------------------------------------------------

typedef __attribute__((ext_vector_type(8)))  __bf16 bf16x8;
typedef __attribute__((ext_vector_type(16))) __bf16 bf16x16;
typedef __attribute__((ext_vector_type(8)))  float  f32x8;

#define NW 8            // 8 waves per 256-thread block (wave32)
#define LN_EPS 1e-5f
#define SCALE_HD32 0.17677669529663687f   // 1/sqrt(32)

// A fragment (16x32 bf16): lane L holds row M=L%16;
//   K = (L>=16 ? 8:0) + e   for e in 0..7   (16B chunk)
//   K = 16 + (L>=16?8:0)+e-8 for e in 8..15 (16B chunk at +16 elements)
__device__ __forceinline__ bf16x16 ldfrag_a(const __bf16* A, int ld) {
  const int lane = threadIdx.x & 31;
  const __bf16* p = A + (lane & 15) * ld + ((lane >> 4) << 3);
  union { bf16x16 v; bf16x8 h[2]; } u;
  u.h[0] = *(const bf16x8*)(p);
  u.h[1] = *(const bf16x8*)(p + 16);
  return u.v;
}

// B fragment (32x16 bf16) loaded from row-major Bt (N rows of K):
//   lane L -> N=L%16 ; same K order as A: K = (L>=16?8:0) + e (e<8), +16 for e>=8
__device__ __forceinline__ bf16x16 ldfrag_b(const __bf16* Bt, int ld) {
  const int lane = threadIdx.x & 31;
  const __bf16* p = Bt + (lane & 15) * ld + ((lane >> 4) << 3);
  union { bf16x16 v; bf16x8 h[2]; } u;
  u.h[0] = *(const bf16x8*)(p);
  u.h[1] = *(const bf16x8*)(p + 16);
  return u.v;
}

// 16x16 output tile: C = A(16xK) @ Bt(16xK)^T, f32 accumulate.
__device__ __forceinline__ f32x8 tile_gemm(const __bf16* A, int lda,
                                           const __bf16* Bt, int ldb, int K) {
  f32x8 acc = {0.f, 0.f, 0.f, 0.f, 0.f, 0.f, 0.f, 0.f};
  for (int k = 0; k < K; k += 32) {
    bf16x16 a = ldfrag_a(A + k, lda);
    bf16x16 b = ldfrag_b(Bt + k, ldb);
    acc = __builtin_amdgcn_wmma_f32_16x16x32_bf16(false, a, false, b,
                                                  (short)0, acc, false, false);
  }
  return acc;
}

// LayerNorm over D=256 per token -> bf16 dst. One wave per token, 8 tokens/pass.
__device__ __forceinline__ void ln_tokens(const float* src, long stride, int ntok,
                                          const float* g, const float* b,
                                          __bf16* dst, int ldd) {
  const int lane = threadIdx.x & 31;
  const int sub  = threadIdx.x >> 5;
  for (int t0 = 0; t0 < ntok; t0 += NW) {
    const int t = t0 + sub;
    const float* p = src + (long)t * stride;
    float v[8]; float s = 0.f;
#pragma unroll
    for (int i = 0; i < 8; ++i) { v[i] = p[lane + 32 * i]; s += v[i]; }
#pragma unroll
    for (int off = 16; off > 0; off >>= 1) s += __shfl_xor(s, off, 32);
    const float mean = s * (1.f / 256.f);
    float q = 0.f;
#pragma unroll
    for (int i = 0; i < 8; ++i) { const float d = v[i] - mean; q += d * d; }
#pragma unroll
    for (int off = 16; off > 0; off >>= 1) q += __shfl_xor(q, off, 32);
    const float rstd = rsqrtf(q * (1.f / 256.f) + LN_EPS);
#pragma unroll
    for (int i = 0; i < 8; ++i) {
      const int c = lane + 32 * i;
      dst[t * ldd + c] = (__bf16)((v[i] - mean) * rstd * g[c] + b[c]);
    }
  }
}

// In-place softmax over a bf16 score row (same index read->write per thread).
__device__ __forceinline__ void softmax_row(__bf16* row, int n) {
  float mx = -3.4e38f;
  for (int i = 0; i < n; ++i) mx = fmaxf(mx, (float)row[i]);
  float s = 0.f;
  for (int i = 0; i < n; ++i) {
    const float e = __expf((float)row[i] - mx);
    s += e; row[i] = (__bf16)e;
  }
  const float inv = 1.f / s;
  for (int i = 0; i < n; ++i) row[i] = (__bf16)((float)row[i] * inv);
}

// ---------------------------------------------------------------------------
__global__ void cvt_bf16_kernel(const float* __restrict__ src,
                                __bf16* __restrict__ dst, int n) {
  const int i = blockIdx.x * 256 + threadIdx.x;
  if (i < n) dst[i] = (__bf16)src[i];
}

// ---------------------------------------------------------------------------
// Phase 1: feature attention over FN=64 per (b,s) row. grid = B*S = 1280.
__global__ __launch_bounds__(256)
void feat_attn_kernel(const float* __restrict__ x, const float* __restrict__ mask,
                      const __bf16* Wq, const __bf16* Wk, const __bf16* Wv,
                      const __bf16* Wo,
                      const float* bq, const float* bk, const float* bv,
                      const float* bo,
                      const float* g, const float* bln,
                      float* __restrict__ X1) {
  __shared__ __attribute__((aligned(16))) __bf16 s_xn[64 * 256];
  __shared__ __attribute__((aligned(16))) __bf16 s_q [64 * 256];
  __shared__ __attribute__((aligned(16))) __bf16 s_k [64 * 256];
  __shared__ __attribute__((aligned(16))) __bf16 s_vt[256 * 64]; // V transposed
  __shared__ __attribute__((aligned(16))) __bf16 s_p [64 * 64];  // scores/probs
  __shared__ __attribute__((aligned(16))) __bf16 s_o [64 * 256];

  const long base = (long)blockIdx.x * (64 * 256);
  const int wave = threadIdx.x >> 5, lane = threadIdx.x & 31;
  const int l15 = lane & 15, hi8 = (lane >> 4) << 3;

  __builtin_prefetch(Wq, 0, 1);
  __builtin_prefetch(Wk, 0, 1);
  __builtin_prefetch(Wv, 0, 1);

  ln_tokens(x + base, 256, 64, g, bln, s_xn, 256);
  __syncthreads();

  // Q,K,V projections: M=64,N=256,K=256 -> 64 tiles / 8 waves.
  for (int t = wave; t < 64; t += NW) {
    const int m0 = (t >> 4) << 4, n0 = (t & 15) << 4;
    f32x8 cq = tile_gemm(s_xn + m0 * 256, 256, Wq + n0 * 256, 256, 256);
    f32x8 ck = tile_gemm(s_xn + m0 * 256, 256, Wk + n0 * 256, 256, 256);
    f32x8 cv = tile_gemm(s_xn + m0 * 256, 256, Wv + n0 * 256, 256, 256);
    const int n = n0 + l15;
#pragma unroll
    for (int r = 0; r < 8; ++r) {
      const int m = m0 + r + hi8;
      s_q[m * 256 + n]  = (__bf16)(cq[r] + bq[n]);
      s_k[m * 256 + n]  = (__bf16)(ck[r] + bk[n]);
      s_vt[n * 64 + m]  = (__bf16)(cv[r] + bv[n]);
    }
  }
  __syncthreads();

  for (int h = 0; h < 8; ++h) {
    // scores S = Q_h @ K_h^T : 16 tiles
    for (int t = wave; t < 16; t += NW) {
      const int m0 = (t >> 2) << 4, n0 = (t & 3) << 4;
      f32x8 c = tile_gemm(s_q + m0 * 256 + h * 32, 256,
                          s_k + n0 * 256 + h * 32, 256, 32);
      const int n = n0 + l15;
#pragma unroll
      for (int r = 0; r < 8; ++r) {
        const int m = m0 + r + hi8;
        float v = c[r] * SCALE_HD32;
        if (h < 4 && mask[m * 64 + n] != 1.0f) v = -3.0e38f; // additive -inf
        s_p[m * 64 + n] = (__bf16)v;
      }
    }
    __syncthreads();
    if (threadIdx.x < 64) softmax_row(s_p + threadIdx.x * 64, 64);
    __syncthreads();
    // O_h = P @ V_h : 8 tiles (heads write disjoint 32-col slices of s_o)
    for (int t = wave; t < 8; t += NW) {
      const int m0 = (t >> 1) << 4, n0 = (t & 1) << 4;
      f32x8 c = tile_gemm(s_p + m0 * 64, 64, s_vt + (h * 32 + n0) * 64, 64, 64);
      const int n = n0 + l15;
#pragma unroll
      for (int r = 0; r < 8; ++r)
        s_o[(m0 + r + hi8) * 256 + h * 32 + n] = (__bf16)c[r];
    }
    __syncthreads();
  }

  // out-proj + residual -> X1
  for (int t = wave; t < 64; t += NW) {
    const int m0 = (t >> 4) << 4, n0 = (t & 15) << 4;
    f32x8 c = tile_gemm(s_o + m0 * 256, 256, Wo + n0 * 256, 256, 256);
    const int n = n0 + l15;
#pragma unroll
    for (int r = 0; r < 8; ++r) {
      const long idx = base + (long)(m0 + r + hi8) * 256 + n;
      X1[idx] = x[idx] + c[r] + bo[n];
    }
  }
}

// ---------------------------------------------------------------------------
// Phase 2: column attention (self for train, cross for test).
// grid.x = B*FN = 256 rows, grid.y = query tiles of 64 tokens.
__global__ __launch_bounds__(256)
void col_attn_kernel(const float* __restrict__ qbase,
                     const float* __restrict__ kvbase,
                     float* __restrict__ outbase, int q_s0,
                     const float* gq, const float* bqln,
                     const float* gkv, const float* bkvln,
                     const __bf16* Wq, const __bf16* Wk, const __bf16* Wv,
                     const __bf16* Wo,
                     const float* bq, const float* bk, const float* bv,
                     const float* bo) {
  __shared__ __attribute__((aligned(16))) __bf16 s_kvn[256 * 256]; // LN'd KV
  __shared__ __attribute__((aligned(16))) __bf16 s_en [64 * 256];  // LN'd Q src
  __shared__ __attribute__((aligned(16))) __bf16 s_qh [64 * 32];
  __shared__ __attribute__((aligned(16))) __bf16 s_kh [256 * 32];
  __shared__ __attribute__((aligned(16))) __bf16 s_vth[32 * 256];  // V_h^T
  __shared__ __attribute__((aligned(16))) __bf16 s_p  [64 * 256];  // scores
  __shared__ __attribute__((aligned(16))) __bf16 s_o  [64 * 256];

  const int rrow = blockIdx.x;
  const int b = rrow >> 6, f = rrow & 63;
  const long rowoff = ((long)b * (320 * 64) + f) * 256;
  const long SSTR = 64 * 256;   // seq stride in elements
  const float* qt  = qbase  + rowoff + (long)(q_s0 + blockIdx.y * 64) * SSTR;
  const float* kvt = kvbase + rowoff;
  float* outt      = outbase + rowoff + (long)(q_s0 + blockIdx.y * 64) * SSTR;

  const int wave = threadIdx.x >> 5, lane = threadIdx.x & 31;
  const int l15 = lane & 15, hi8 = (lane >> 4) << 3;

  __builtin_prefetch(Wq, 0, 1);
  __builtin_prefetch(Wk, 0, 1);

  ln_tokens(kvt, SSTR, 256, gkv, bkvln, s_kvn, 256);
  ln_tokens(qt,  SSTR, 64,  gq,  bqln,  s_en,  256);
  __syncthreads();

  for (int h = 0; h < 8; ++h) {
    const __bf16* wqh = Wq + (long)h * 32 * 256;
    const __bf16* wkh = Wk + (long)h * 32 * 256;
    const __bf16* wvh = Wv + (long)h * 32 * 256;
    // Q_h : 8 tiles
    for (int t = wave; t < 8; t += NW) {
      const int m0 = (t >> 1) << 4, n0 = (t & 1) << 4;
      f32x8 c = tile_gemm(s_en + m0 * 256, 256, wqh + n0 * 256, 256, 256);
      const int n = n0 + l15;
#pragma unroll
      for (int r = 0; r < 8; ++r)
        s_qh[(m0 + r + hi8) * 32 + n] = (__bf16)(c[r] + bq[h * 32 + n]);
    }
    // K_h, V_h (V stored transposed) : 32 tiles
    for (int t = wave; t < 32; t += NW) {
      const int m0 = (t >> 1) << 4, n0 = (t & 1) << 4;
      f32x8 ck = tile_gemm(s_kvn + m0 * 256, 256, wkh + n0 * 256, 256, 256);
      f32x8 cv = tile_gemm(s_kvn + m0 * 256, 256, wvh + n0 * 256, 256, 256);
      const int n = n0 + l15;
#pragma unroll
      for (int r = 0; r < 8; ++r) {
        const int m = m0 + r + hi8;
        s_kh[m * 32 + n]   = (__bf16)(ck[r] + bk[h * 32 + n]);
        s_vth[n * 256 + m] = (__bf16)(cv[r] + bv[h * 32 + n]);
      }
    }
    __syncthreads();
    // S = Q_h @ K_h^T : 64 tiles
    for (int t = wave; t < 64; t += NW) {
      const int m0 = (t >> 4) << 4, n0 = (t & 15) << 4;
      f32x8 c = tile_gemm(s_qh + m0 * 32, 32, s_kh + n0 * 32, 32, 32);
      const int n = n0 + l15;
#pragma unroll
      for (int r = 0; r < 8; ++r)
        s_p[(m0 + r + hi8) * 256 + n] = (__bf16)(c[r] * SCALE_HD32);
    }
    __syncthreads();
    if (threadIdx.x < 64) softmax_row(s_p + threadIdx.x * 256, 256);
    __syncthreads();
    // O_h = P @ V_h : 8 tiles -> disjoint column slice of s_o
    for (int t = wave; t < 8; t += NW) {
      const int m0 = (t >> 1) << 4, n0 = (t & 1) << 4;
      f32x8 c = tile_gemm(s_p + m0 * 256, 256, s_vth + n0 * 256, 256, 256);
      const int n = n0 + l15;
#pragma unroll
      for (int r = 0; r < 8; ++r)
        s_o[(m0 + r + hi8) * 256 + h * 32 + n] = (__bf16)c[r];
    }
    __syncthreads();
  }

  // out-proj + residual -> outbase
  for (int t = wave; t < 64; t += NW) {
    const int m0 = (t >> 4) << 4, n0 = (t & 15) << 4;
    f32x8 c = tile_gemm(s_o + m0 * 256, 256, Wo + n0 * 256, 256, 256);
    const int n = n0 + l15;
#pragma unroll
    for (int r = 0; r < 8; ++r) {
      const int m = m0 + r + hi8;
      outt[(long)m * SSTR + n] = qt[(long)m * SSTR + n] + c[r] + bo[n];
    }
  }
}

// ---------------------------------------------------------------------------
// Phase 3: SwiGLU MLP, in-place on d_out. grid = 81920/64 = 1280.
__global__ __launch_bounds__(256)
void mlp_kernel(float* __restrict__ out, const float* gm, const float* bm,
                const __bf16* W1, const float* b1,
                const __bf16* Wg, const float* bg,
                const __bf16* W2, const float* b2) {
  __shared__ __attribute__((aligned(16))) __bf16 s_xn[64 * 256];
  __shared__ __attribute__((aligned(16))) __bf16 s_h [64 * 1024];

  const long base = (long)blockIdx.x * (64 * 256);
  const int wave = threadIdx.x >> 5, lane = threadIdx.x & 31;
  const int l15 = lane & 15, hi8 = (lane >> 4) << 3;

  __builtin_prefetch(W1, 0, 1);
  __builtin_prefetch(Wg, 0, 1);

  ln_tokens(out + base, 256, 64, gm, bm, s_xn, 256);
  __syncthreads();

  // h = (xn W1^T + b1) * silu(xn Wg^T + bg) : 256 tiles (M=64,N=1024)
  for (int t = wave; t < 256; t += NW) {
    const int m0 = (t >> 6) << 4, n0 = (t & 63) << 4;
    f32x8 c1 = tile_gemm(s_xn + m0 * 256, 256, W1 + (long)n0 * 256, 256, 256);
    f32x8 cg = tile_gemm(s_xn + m0 * 256, 256, Wg + (long)n0 * 256, 256, 256);
    const int n = n0 + l15;
#pragma unroll
    for (int r = 0; r < 8; ++r) {
      const float a  = c1[r] + b1[n];
      const float gv = cg[r] + bg[n];
      const float sg = gv / (1.0f + __expf(-gv));       // silu
      s_h[(m0 + r + hi8) * 1024 + n] = (__bf16)(a * sg);
    }
  }
  __syncthreads();

  // delta = h W2^T + b2 ; out += delta : 64 tiles (M=64,N=256,K=1024)
  for (int t = wave; t < 64; t += NW) {
    const int m0 = (t >> 4) << 4, n0 = (t & 15) << 4;
    f32x8 c = tile_gemm(s_h + m0 * 1024, 1024, W2 + (long)n0 * 1024, 1024, 1024);
    const int n = n0 + l15;
#pragma unroll
    for (int r = 0; r < 8; ++r) {
      const long idx = base + (long)(m0 + r + hi8) * 256 + n;
      out[idx] = out[idx] + c[r] + b2[n];
    }
  }
}

// ---------------------------------------------------------------------------
extern "C" void kernel_launch(void* const* d_in, const int* in_sizes, int n_in,
                              void* d_out, int out_size, void* d_ws, size_t ws_size,
                              hipStream_t stream) {
  (void)in_sizes; (void)n_in; (void)out_size; (void)ws_size;

  const float* x    = (const float*)d_in[0];
  const float* mask = (const float*)d_in[1];

  // ws layout: X1 (phase-1 output, fp32) then bf16 weights.
  const size_t NTOT = (size_t)4 * 320 * 64 * 256;   // 20,971,520 elems
  float*  X1 = (float*)d_ws;
  __bf16* wb = (__bf16*)((char*)d_ws + NTOT * sizeof(float));
  __bf16* Wb[12];
  for (int i = 0; i < 12; ++i) Wb[i] = wb + (size_t)i * 65536;
  __bf16* W1b = wb + (size_t)12 * 65536;
  __bf16* Wgb = W1b + 262144;
  __bf16* W2b = Wgb + 262144;

  // fp32 -> bf16 weight conversion
  static const int widx[12] = {2, 3, 4, 5, 12, 13, 14, 15, 22, 23, 24, 25};
  for (int i = 0; i < 12; ++i)
    cvt_bf16_kernel<<<256, 256, 0, stream>>>((const float*)d_in[widx[i]], Wb[i], 65536);
  cvt_bf16_kernel<<<1024, 256, 0, stream>>>((const float*)d_in[32], W1b, 262144);
  cvt_bf16_kernel<<<1024, 256, 0, stream>>>((const float*)d_in[34], Wgb, 262144);
  cvt_bf16_kernel<<<1024, 256, 0, stream>>>((const float*)d_in[36], W2b, 262144);

  // Phase 1: feature attention -> X1
  feat_attn_kernel<<<1280, 256, 0, stream>>>(
      x, mask, Wb[0], Wb[1], Wb[2], Wb[3],
      (const float*)d_in[6], (const float*)d_in[7],
      (const float*)d_in[8], (const float*)d_in[9],
      (const float*)d_in[10], (const float*)d_in[11], X1);

  // Phase 2a: train self-attention (q & kv from X1) -> d_out train tokens
  col_attn_kernel<<<dim3(256, 4), 256, 0, stream>>>(
      X1, X1, (float*)d_out, /*q_s0=*/0,
      (const float*)d_in[20], (const float*)d_in[21],
      (const float*)d_in[20], (const float*)d_in[21],
      Wb[4], Wb[5], Wb[6], Wb[7],
      (const float*)d_in[16], (const float*)d_in[17],
      (const float*)d_in[18], (const float*)d_in[19]);

  // Phase 2b: test cross-attention (q from X1 test, kv from updated train in
  // d_out) -> d_out test tokens
  col_attn_kernel<<<dim3(256, 1), 256, 0, stream>>>(
      X1, (const float*)d_out, (float*)d_out, /*q_s0=*/256,
      (const float*)d_in[30], (const float*)d_in[31],
      (const float*)d_in[30], (const float*)d_in[31],
      Wb[8], Wb[9], Wb[10], Wb[11],
      (const float*)d_in[26], (const float*)d_in[27],
      (const float*)d_in[28], (const float*)d_in[29]);

  // Phase 3: SwiGLU MLP, in-place residual on d_out
  mlp_kernel<<<1280, 256, 0, stream>>>(
      (float*)d_out, (const float*)d_in[38], (const float*)d_in[39],
      W1b, (const float*)d_in[33], Wgb, (const float*)d_in[35],
      W2b, (const float*)d_in[37]);
}
